// SABlock_54949811585193
// MI455X (gfx1250) — compile-verified, matrix operand order checked by reading, not run on hardware
//
#include <hip/hip_runtime.h>
#include <hip/hip_bf16.h>

typedef _Float16 half_t;
typedef __attribute__((ext_vector_type(16))) _Float16 v16h;
typedef __attribute__((ext_vector_type(8)))  float    v8f;

#define NB  8
#define NS  1024
#define NC  768
#define NH  12
#define NHD 64
#define NBH 96
#define NQH 32

// ---------------------------------------------------------------------------
// CDNA5 feature guards (probe-verified builtins where possible)
// ---------------------------------------------------------------------------
#if __has_builtin(__builtin_amdgcn_global_load_async_to_lds_b128) && \
    __has_builtin(__builtin_amdgcn_s_wait_asynccnt)
#define HAVE_ASYNC 1
#else
#define HAVE_ASYNC 0
#endif

#if __has_builtin(__builtin_amdgcn_tensor_load_to_lds) && \
    __has_builtin(__builtin_amdgcn_s_wait_tensorcnt)
#define HAVE_TDM 1
#else
#define HAVE_TDM 0
#endif

#if HAVE_ASYNC
typedef int i32x4v __attribute__((ext_vector_type(4)));
__device__ __forceinline__ void async_b128(const void* g, void* l) {
    // Signature per hipcc diagnostic: (global v4i*, local v4i*, imm, imm)
    __builtin_amdgcn_global_load_async_to_lds_b128(
        (__attribute__((address_space(1))) i32x4v*)(g),
        (__attribute__((address_space(3))) i32x4v*)(l), 0, 0);
}
__device__ __forceinline__ void wait_async() {
    __builtin_amdgcn_s_wait_asynccnt(0);
}
#endif

#if HAVE_TDM
typedef uint32_t u32x4_t __attribute__((ext_vector_type(4)));
typedef int      i32x4_t __attribute__((ext_vector_type(4)));
typedef int      i32x8_t __attribute__((ext_vector_type(8)));

// 2-D f16 tile DMA: rows x rowlen elements, row stride = stride_elems,
// packed contiguously into LDS at byte offset lds_off.
// D# packing per CDNA5 ISA.
__device__ __forceinline__ void tdm_load_tile_f16(
    uint32_t lds_off, const half_t* gptr, int rows, int rowlen, int stride_elems)
{
    uint64_t ga = (uint64_t)(uintptr_t)gptr;
    u32x4_t g0;
    g0[0] = 1u;                                   // count=1, user mode
    g0[1] = lds_off;                              // lds_addr
    g0[2] = (uint32_t)ga;                         // global_addr[31:0]
    g0[3] = (uint32_t)(ga >> 32) | (2u << 30);    // global_addr[56:32] | type=2
    i32x8_t g1;
    g1[0] = (1 << 16);                            // data_size=1 (2 bytes)
    g1[1] = (int)((rowlen & 0xffff) << 16);       // tensor_dim0[15:0]
    g1[2] = (int)(((unsigned)rowlen >> 16) | ((rows & 0xffff) << 16));   // dim0 hi | dim1 lo
    g1[3] = (int)((((unsigned)rows >> 16) & 0xffffu) | ((rowlen & 0xffff) << 16)); // dim1 hi | tile_dim0
    g1[4] = (int)(rows & 0xffff);                 // tile_dim1 | tile_dim2=0
    g1[5] = stride_elems;                         // tensor_dim0_stride[31:0]
    g1[6] = 0;
    g1[7] = 0;
    i32x4_t z4 = {0, 0, 0, 0};
#if __clang_major__ >= 23
    i32x8_t z8 = {0, 0, 0, 0, 0, 0, 0, 0};
    __builtin_amdgcn_tensor_load_to_lds(g0, g1, z4, z4, z8, 0);
#else
    __builtin_amdgcn_tensor_load_to_lds(g0, g1, z4, z4, 0);
#endif
}
__device__ __forceinline__ uint32_t lds_offset_of(const void* p) {
    return (uint32_t)(uintptr_t)(__attribute__((address_space(3))) void*)(p);
}
#endif

// ---------------------------------------------------------------------------
// WMMA 16x16x32 f16 A/B fragment layout: lanes 0-15 hold K {0..7,16..23}
// (as f16 pairs), lanes 16-31 hold K {8..15,24..31}; lane&15 = row (M or N).
// ---------------------------------------------------------------------------
__device__ __forceinline__ v16h frag_from_rows32(const half_t* base, int l15, int hi) {
    const half_t* p = base + l15 * 32 + hi * 8;
    union { v16h h; uint4 u[2]; } f;
    f.u[0] = *(const uint4*)(p);
    f.u[1] = *(const uint4*)(p + 16);
    return f.h;
}

__device__ __forceinline__ v16h frag_from_global64(const half_t* rowptr, int c, int hi) {
    const half_t* p = rowptr + c * 32 + hi * 8;
    union { v16h h; uint4 u[2]; } f;
    f.u[0] = *(const uint4*)(p);
    f.u[1] = *(const uint4*)(p + 16);
    return f.h;
}

// ---------------------------------------------------------------------------
// Elementwise f32 -> f16 conversion (one-time; halves HBM traffic for GEMMs)
// ---------------------------------------------------------------------------
__global__ __launch_bounds__(256) void cvt_f32_f16_kernel(
    const float* __restrict__ in, half_t* __restrict__ outp, int n4)
{
    const int i = (blockIdx.x * 256 + threadIdx.x) * 4;
    if (i < n4 * 4) {
        float4 f = *(const float4*)(in + i);
        outp[i + 0] = (half_t)f.x;
        outp[i + 1] = (half_t)f.y;
        outp[i + 2] = (half_t)f.z;
        outp[i + 3] = (half_t)f.w;
    }
}

// ---------------------------------------------------------------------------
// Tiled GEMM  C[M,N] = A[M,K] @ B[N,K]^T, f16 in / f32 accum via WMMA.
// Block tile 128x128, 256 threads = 8 waves (4 M x 2 N), wave tile 32x64.
// K stepped by 32; A/B tiles staged in LDS, double-buffered, loaded by the
// Tensor Data Mover when available (wave 0 issues, TENSORcnt + barrier sync).
// MODE 0: scatter to packed f16 q/k/v (B,H,S,HD). MODE 1: f32 out + bias.
// ---------------------------------------------------------------------------
template <int MODE>
__global__ __launch_bounds__(256) void gemm_wmma_kernel(
    const half_t* __restrict__ A, const half_t* __restrict__ Bw,
    const float* __restrict__ bias, void* __restrict__ Cout,
    int M, int N, int K)
{
    __shared__ half_t As[2][128 * 32];
    __shared__ half_t Bs[2][128 * 32];

    const int t     = threadIdx.x;
    const int lane  = t & 31;
    const int wid   = t >> 5;
    const int mbase = (wid & 3) * 32;
    const int nbase = (wid >> 2) * 64;
    const int l15   = lane & 15;
    const int hi    = lane >> 4;
    const int blockM = blockIdx.y * 128;
    const int blockN = blockIdx.x * 128;

    v8f acc[2][4];
    for (int i = 0; i < 2; ++i)
        for (int j = 0; j < 4; ++j)
            for (int e = 0; e < 8; ++e) acc[i][j][e] = 0.0f;

    const int nk = K / 32;

#if HAVE_TDM
    if (wid == 0) {
        tdm_load_tile_f16(lds_offset_of(&As[0][0]), A  + (size_t)blockM * K, 128, 32, K);
        tdm_load_tile_f16(lds_offset_of(&Bs[0][0]), Bw + (size_t)blockN * K, 128, 32, K);
        __builtin_amdgcn_s_wait_tensorcnt(0);
    }
#else
    {
        for (int c = 0; c < 2; ++c) {
            const int idx = (t + c * 256) * 8;          // f16 index in 128x32 tile
            const int row = idx >> 5, col = idx & 31;
            *(uint4*)&As[0][idx] = *(const uint4*)(A  + (size_t)(blockM + row) * K + col);
            *(uint4*)&Bs[0][idx] = *(const uint4*)(Bw + (size_t)(blockN + row) * K + col);
        }
    }
#endif
    __syncthreads();

    for (int kc = 0; kc < nk; ++kc) {
        const int buf = kc & 1;

        if (kc + 1 < nk) {
            const int k0 = (kc + 1) * 32;
#if HAVE_TDM
            if (wid == 0) {
                tdm_load_tile_f16(lds_offset_of(&As[buf ^ 1][0]),
                                  A  + (size_t)blockM * K + k0, 128, 32, K);
                tdm_load_tile_f16(lds_offset_of(&Bs[buf ^ 1][0]),
                                  Bw + (size_t)blockN * K + k0, 128, 32, K);
            }
#else
            for (int c = 0; c < 2; ++c) {
                const int idx = (t + c * 256) * 8;
                const int row = idx >> 5, col = idx & 31;
                *(uint4*)&As[buf ^ 1][idx] =
                    *(const uint4*)(A  + (size_t)(blockM + row) * K + k0 + col);
                *(uint4*)&Bs[buf ^ 1][idx] =
                    *(const uint4*)(Bw + (size_t)(blockN + row) * K + k0 + col);
            }
#endif
        }

        v16h af[2], bf[4];
        for (int i = 0; i < 2; ++i)
            af[i] = frag_from_rows32(&As[buf][(mbase + 16 * i) * 32], l15, hi);
        for (int j = 0; j < 4; ++j)
            bf[j] = frag_from_rows32(&Bs[buf][(nbase + 16 * j) * 32], l15, hi);

        for (int i = 0; i < 2; ++i)
            for (int j = 0; j < 4; ++j)
                acc[i][j] = __builtin_amdgcn_wmma_f32_16x16x32_f16(
                    false, af[i], false, bf[j], (short)0, acc[i][j], false, false);

#if HAVE_TDM
        if (wid == 0 && kc + 1 < nk) __builtin_amdgcn_s_wait_tensorcnt(0);
#endif
        __syncthreads();
    }

    // Epilogue. D layout: element e: row = e + 8*hi, col = lane&15.
    for (int i = 0; i < 2; ++i) {
        for (int j = 0; j < 4; ++j) {
            for (int e = 0; e < 8; ++e) {
                const int grow = blockM + mbase + 16 * i + e + 8 * hi;
                const int gcol = blockN + nbase + 16 * j + l15;
                const float val = acc[i][j][e];
                if (MODE == 0) {
                    half_t* qkv = (half_t*)Cout;
                    const int part = gcol / NC;
                    const int rem  = gcol - part * NC;
                    const int hh   = rem >> 6;
                    const int d    = rem & 63;
                    const int b    = grow >> 10;
                    const int s    = grow & 1023;
                    qkv[(size_t)part * ((size_t)NBH * NS * NHD)
                        + (((size_t)b * NH + hh) * NS + s) * NHD + d] = (half_t)val;
                } else {
                    float* o = (float*)Cout;
                    o[(size_t)grow * N + gcol] = val + bias[gcol];
                }
            }
        }
    }
}

// ---------------------------------------------------------------------------
// Decomposed relative position bias as 32x32x64 WMMA GEMMs.
// mode 0: rel_h[bh, qh, qw, kh] = q[bh, qh*32+qw] . rph[qh-kh+31]
// mode 1: rel_w[bh, qh, qw, kw] = q[bh, qh*32+qw] . rpw[qw-kw+31]
// One wave per block; output [bh][s][k] f32.
// ---------------------------------------------------------------------------
__global__ __launch_bounds__(32) void rel_bias_kernel(
    const half_t* __restrict__ q, const float* __restrict__ rph,
    const float* __restrict__ rpw, float* __restrict__ relH,
    float* __restrict__ relW)
{
    const int fixed = blockIdx.x;
    const int bh    = blockIdx.y;
    const int mode  = blockIdx.z;
    const int lane  = threadIdx.x;
    const int l15   = lane & 15;
    const int hi    = lane >> 4;
    const float* rp = mode ? rpw : rph;

    v8f acc[2][2];
    for (int i = 0; i < 2; ++i)
        for (int j = 0; j < 2; ++j)
            for (int e = 0; e < 8; ++e) acc[i][j][e] = 0.0f;

    for (int kc = 0; kc < 2; ++kc) {
        v16h af[2], bf[2];
        for (int i = 0; i < 2; ++i) {
            const int rr = i * 16 + l15;
            const int s  = mode ? (rr * NQH + fixed) : (fixed * NQH + rr);
            af[i] = frag_from_global64(q + ((size_t)bh * NS + s) * NHD, kc, hi);
        }
        for (int j = 0; j < 2; ++j) {
            const int kk   = j * 16 + l15;
            const int rrow = fixed - kk + (NQH - 1);
            const float* bp = rp + (size_t)rrow * NHD + kc * 32 + hi * 8;
            v16h hv;
            for (int e = 0; e < 8; ++e) hv[e]     = (half_t)bp[e];
            for (int e = 0; e < 8; ++e) hv[8 + e] = (half_t)bp[16 + e];
            bf[j] = hv;
        }
        for (int i = 0; i < 2; ++i)
            for (int j = 0; j < 2; ++j)
                acc[i][j] = __builtin_amdgcn_wmma_f32_16x16x32_f16(
                    false, af[i], false, bf[j], (short)0, acc[i][j], false, false);
    }

    float* dst = mode ? relW : relH;
    for (int i = 0; i < 2; ++i) {
        for (int j = 0; j < 2; ++j) {
            for (int e = 0; e < 8; ++e) {
                const int row = i * 16 + e + 8 * hi;
                const int col = j * 16 + l15;
                const int s   = mode ? (row * NQH + fixed) : (fixed * NQH + row);
                dst[((size_t)bh * NS + s) * NQH + col] = acc[i][j][e];
            }
        }
    }
}

// ---------------------------------------------------------------------------
// Flash attention with decomposed rel bias.
// Grid: (S/64, B*H). Block: 128 threads = 4 waves; each wave owns 16 query
// rows. Streams 32-key chunks; softmax state and O stay in registers;
// P and V bounce through LDS to form WMMA fragments. rel slices arrive via
// async global->LDS loads (ASYNCcnt) when the toolchain supports them.
// ---------------------------------------------------------------------------
__global__ __launch_bounds__(128) void attn_kernel(
    const half_t* __restrict__ q, const half_t* __restrict__ k,
    const half_t* __restrict__ v, const float* __restrict__ relH,
    const float* __restrict__ relW, half_t* __restrict__ aout)
{
    __shared__ float  relHs[64 * 32];
    __shared__ float  relWs[64 * 32];
    __shared__ half_t Vt[64 * 32];        // [d][key] transposed V chunk
    __shared__ half_t Pl[4][16 * 32];     // per-wave P staging

    const int bh    = blockIdx.y;
    const int b     = bh / NH;
    const int h     = bh - b * NH;
    const int qbase = blockIdx.x * 64;
    const int t     = threadIdx.x;
    const int lane  = t & 31;
    const int w     = t >> 5;
    const int l15   = lane & 15;
    const int hi    = lane >> 4;

    {
        const float* srcH = relH + ((size_t)bh * NS + qbase) * NQH;
        const float* srcW = relW + ((size_t)bh * NS + qbase) * NQH;
        for (int i = t * 4; i < 64 * 32; i += 128 * 4) {
#if HAVE_ASYNC
            async_b128(srcH + i, &relHs[i]);
            async_b128(srcW + i, &relWs[i]);
#else
            *(float4*)&relHs[i] = *(const float4*)(srcH + i);
            *(float4*)&relWs[i] = *(const float4*)(srcW + i);
#endif
        }
    }

    const int qrow = qbase + w * 16 + l15;
    const half_t* qp = q + ((size_t)bh * NS + qrow) * NHD;
    v16h Qa[2];
    Qa[0] = frag_from_global64(qp, 0, hi);
    Qa[1] = frag_from_global64(qp, 1, hi);

    float m_r[8], l_r[8];
    v8f O[4];
    for (int e = 0; e < 8; ++e) { m_r[e] = -1.0e30f; l_r[e] = 0.0f; }
    for (int dt = 0; dt < 4; ++dt)
        for (int e = 0; e < 8; ++e) O[dt][e] = 0.0f;

    int qloc[8];
    for (int e = 0; e < 8; ++e) qloc[e] = w * 16 + e + 8 * hi;

#if HAVE_ASYNC
    wait_async();
#endif
    __syncthreads();

    const int vk = t >> 2;
    const int vd = (t & 3) * 16;

    for (int kc = 0; kc < NS / 32; ++kc) {
        const int kbase = kc * 32;

        __syncthreads();   // previous chunk's Vt reads complete
        {   // cooperative transposed V stage: Vt[d][key]
            const half_t* vp = v + ((size_t)bh * NS + kbase + vk) * NHD + vd;
            union { uint4 u[2]; half_t s[16]; } buf;
            buf.u[0] = *(const uint4*)(vp);
            buf.u[1] = *(const uint4*)(vp + 8);
            for (int i = 0; i < 16; ++i) Vt[(vd + i) * 32 + vk] = buf.s[i];
            if (kc + 1 < NS / 32) {
                __builtin_prefetch(v + ((size_t)bh * NS + kbase + 32 + vk) * NHD + vd, 0, 3);
                __builtin_prefetch(k + ((size_t)bh * NS + kbase + 32 + vk) * NHD + vd, 0, 3);
            }
        }
        __syncthreads();

        // ---- scores for two 16-key subtiles --------------------------------
        float sc[2][8];
        const int kh = kc;             // key height index, constant per chunk
        float bhv[8];
        for (int e = 0; e < 8; ++e) bhv[e] = relHs[qloc[e] * 32 + kh];

        for (int j = 0; j < 2; ++j) {
            const int key = kbase + j * 16 + l15;
            const half_t* kp = k + ((size_t)bh * NS + key) * NHD;
            v16h Kb0 = frag_from_global64(kp, 0, hi);
            v16h Kb1 = frag_from_global64(kp, 1, hi);
            v8f s;
            for (int e = 0; e < 8; ++e) s[e] = 0.0f;
            s = __builtin_amdgcn_wmma_f32_16x16x32_f16(false, Qa[0], false, Kb0,
                                                       (short)0, s, false, false);
            s = __builtin_amdgcn_wmma_f32_16x16x32_f16(false, Qa[1], false, Kb1,
                                                       (short)0, s, false, false);
            const int kw = j * 16 + l15;
            for (int e = 0; e < 8; ++e)
                sc[j][e] = s[e] * 0.125f + bhv[e] + relWs[qloc[e] * 32 + kw];
        }

        // ---- online softmax ------------------------------------------------
        float mnew[8], corr[8];
        for (int e = 0; e < 8; ++e) {
            float mx = fmaxf(sc[0][e], sc[1][e]);
            for (int off = 1; off < 16; off <<= 1)
                mx = fmaxf(mx, __shfl_xor(mx, off, 32));
            mnew[e] = fmaxf(m_r[e], mx);
            corr[e] = __expf(m_r[e] - mnew[e]);
            m_r[e]  = mnew[e];
        }
        for (int e = 0; e < 8; ++e) {
            sc[0][e] = __expf(sc[0][e] - mnew[e]);
            sc[1][e] = __expf(sc[1][e] - mnew[e]);
            float sum = sc[0][e] + sc[1][e];
            for (int off = 1; off < 16; off <<= 1)
                sum += __shfl_xor(sum, off, 32);
            l_r[e] = l_r[e] * corr[e] + sum;
        }
        for (int dt = 0; dt < 4; ++dt)
            for (int e = 0; e < 8; ++e) O[dt][e] *= corr[e];

        // ---- P -> LDS (f16), reload as A fragment --------------------------
        half_t* pl = &Pl[w][0];
        for (int j = 0; j < 2; ++j)
            for (int e = 0; e < 8; ++e)
                pl[(e + 8 * hi) * 32 + j * 16 + l15] = (half_t)sc[j][e];
        // wave-private LDS region; DS ops from the same wave stay in order
        v16h Pa = frag_from_rows32(pl, l15, hi);

        // ---- O += P @ V -----------------------------------------------------
        for (int dt = 0; dt < 4; ++dt) {
            v16h Vb = frag_from_rows32(&Vt[(dt * 16) * 32], l15, hi);
            O[dt] = __builtin_amdgcn_wmma_f32_16x16x32_f16(
                false, Pa, false, Vb, (short)0, O[dt], false, false);
        }
    }

    // ---- normalize and write (B, S, C) f16 for the output projection -------
    for (int dt = 0; dt < 4; ++dt) {
        const int d = h * NHD + dt * 16 + l15;
        for (int e = 0; e < 8; ++e) {
            const int s = qbase + qloc[e];
            aout[((size_t)b * NS + s) * NC + d] = (half_t)(O[dt][e] / l_r[e]);
        }
    }
}

// ---------------------------------------------------------------------------
extern "C" void kernel_launch(void* const* d_in, const int* in_sizes, int n_in,
                              void* d_out, int out_size, void* d_ws, size_t ws_size,
                              hipStream_t stream) {
    const float* x     = (const float*)d_in[0];
    const float* w_qkv = (const float*)d_in[1];
    const float* w_out = (const float*)d_in[2];
    const float* b_out = (const float*)d_in[3];
    const float* rph   = (const float*)d_in[4];
    const float* rpw   = (const float*)d_in[5];
    float* out = (float*)d_out;

    const size_t nX    = (size_t)NB * NS * NC;       // 6291456
    const size_t nWqkv = (size_t)3 * NC * NC;        // 1769472
    const size_t nWout = (size_t)NC * NC;            // 589824
    const size_t span  = (size_t)NBH * NS * NHD;     // 6291456

    half_t* xh    = (half_t*)d_ws;
    half_t* wqkvh = xh + nX;
    half_t* wouth = wqkvh + nWqkv;
    half_t* qf    = wouth + nWout;
    half_t* kf    = qf + span;
    half_t* vf    = kf + span;
    half_t* aout  = vf + span;
    float*  relH  = (float*)(aout + span);
    float*  relW  = relH + (size_t)NBH * NS * NQH;

    // 0) one-time f32 -> f16 conversions
    cvt_f32_f16_kernel<<<dim3((nX / 4 + 255) / 256), 256, 0, stream>>>(x, xh, (int)(nX / 4));
    cvt_f32_f16_kernel<<<dim3((nWqkv / 4 + 255) / 256), 256, 0, stream>>>(w_qkv, wqkvh, (int)(nWqkv / 4));
    cvt_f32_f16_kernel<<<dim3((nWout / 4 + 255) / 256), 256, 0, stream>>>(w_out, wouth, (int)(nWout / 4));

    // 1) QKV projection: (8192 x 768) @ (2304 x 768)^T, scatter to q/k/v f16
    gemm_wmma_kernel<0><<<dim3(2304 / 128, 8192 / 128), 256, 0, stream>>>(
        xh, wqkvh, nullptr, (void*)qf, NB * NS, 3 * NC, NC);

    // 2) decomposed relative position bias tables
    rel_bias_kernel<<<dim3(NQH, NBH, 2), 32, 0, stream>>>(qf, rph, rpw, relH, relW);

    // 3) flash attention with rel bias
    attn_kernel<<<dim3(NS / 64, NBH), 128, 0, stream>>>(qf, kf, vf, relH, relW, aout);

    // 4) output projection + bias: (8192 x 768) @ (768 x 768)^T + b
    gemm_wmma_kernel<1><<<dim3(768 / 128, 8192 / 128), 256, 0, stream>>>(
        aout, wouth, b_out, (void*)out, NB * NS, NC, NC);
}
